// SpatioTemporalEncoder_27754078667166
// MI455X (gfx1250) — compile-verified, matrix operand order checked by reading, not run on hardware
//
#include <hip/hip_runtime.h>
#include <hip/hip_bf16.h>

// ---------------- problem constants ----------------
#define Bv     4
#define Tv     10
#define Nv     2048
#define Fv     64
#define Hv     128
#define NHEADS 4
#define Dv     32
#define Ev     65536
#define NTOT   (Bv * Nv)        // 8192
#define EF     (Ev + NTOT)      // 73728 edges incl. self loops
#define G4H    (4 * Hv)         // 512

typedef __attribute__((ext_vector_type(16))) __bf16 v16bf;
typedef __attribute__((ext_vector_type(8)))  __bf16 v8bf;
typedef __attribute__((ext_vector_type(8)))  float  v8f;

// ---------------- WMMA helpers (16x16x32 bf16 -> f32) ----------------
__device__ __forceinline__ v8f wmma_bf16(v16bf a, v16bf b, v8f c) {
    return __builtin_amdgcn_wmma_f32_16x16x32_bf16(
        /*neg_a=*/false, a, /*neg_b=*/false, b,
        /*c_mod=*/(short)0, c, /*reuse_a=*/false, /*reuse_b=*/false);
}

// A fragment: 16x32 (MxK), row-major source. ISA layout:
// lanes 0-15 hold M=lane, K = {0..7, 16..23}; lanes 16-31 hold M=lane-16, K = {8..15, 24..31}
__device__ __forceinline__ v16bf load_a_f32(const float* A, int lda, int lane) {
    int row = lane & 15;
    int kb  = (lane & 16) ? 8 : 0;
    const float* p = A + row * lda + kb;
    v16bf r;
#pragma unroll
    for (int j = 0; j < 8; ++j) {
        r[j]     = (__bf16)p[j];
        r[8 + j] = (__bf16)p[j + 16];
    }
    return r;
}

__device__ __forceinline__ v16bf load_a_bf16(const __bf16* A, int lda, int lane) {
    int row = lane & 15;
    int kb  = (lane & 16) ? 8 : 0;
    const __bf16* p = A + row * lda + kb;
    v8bf lo = *(const v8bf*)p;        // 16B aligned (lda=128, kb in {0,8})
    v8bf hi = *(const v8bf*)(p + 16);
    v16bf r;
#pragma unroll
    for (int j = 0; j < 8; ++j) { r[j] = lo[j]; r[8 + j] = hi[j]; }
    return r;
}

// ---------------- ordered-uint float max (for atomic segment max) ----------------
__device__ __forceinline__ unsigned ordf(float f) {
    unsigned u = __float_as_uint(f);
    return (u & 0x80000000u) ? ~u : (u | 0x80000000u);
}
__device__ __forceinline__ float deordf(unsigned u) {
    unsigned v = (u & 0x80000000u) ? (u & 0x7FFFFFFFu) : ~u;
    return __uint_as_float(v);
}
__device__ __forceinline__ float sigm(float x) { return 1.f / (1.f + __expf(-x)); }

// ---------------- weight swizzle: row-major f32 -> WMMA B-fragment bf16 ----------------
// dst layout: [((kt*NT + nt)*32 + lane)*16 + j]; element (k,n):
//   n = nt*16 + (lane&15); k = kt*32 + ((lane>>4)&1)*8 + (j&7) + ((j&8)?16:0)
__global__ void k_swizzle(const float* __restrict__ src, __bf16* __restrict__ dst,
                          int K, int Nn, int transpose) {
    int idx = blockIdx.x * blockDim.x + threadIdx.x;
    if (idx >= K * Nn) return;
    int j    = idx & 15;
    int lane = (idx >> 4) & 31;
    int rest = idx >> 9;
    int NT   = Nn >> 4;
    int nt   = rest % NT;
    int kt   = rest / NT;
    int n = nt * 16 + (lane & 15);
    int k = kt * 32 + ((lane >> 4) & 1) * 8 + (j & 7) + ((j & 8) ? 16 : 0);
    float v = transpose ? src[n * K + k] : src[k * Nn + n];
    dst[idx] = (__bf16)v;
}

__global__ void k_bias_combine(const float* a, const float* b, float* o, int n) {
    int i = blockIdx.x * blockDim.x + threadIdx.x;
    if (i < n) o[i] = a[i] + b[i];
}

// ---------------- GAT: h = x_flat @ W  (x f32 -> bf16 in regs) ----------------
// grid: Tv * (NTOT/16) blocks, 256 threads (8 waves); wave = column tile (H/16 = 8)
__global__ void k_gat_proj(const float* __restrict__ x, const __bf16* __restrict__ Wswz,
                           float* __restrict__ h) {
    int lane = threadIdx.x & 31;
    int wave = threadIdx.x >> 5;
    int tile = blockIdx.x;
    int t       = tile / (NTOT / 16);
    int rowbase = (tile % (NTOT / 16)) * 16;
    int bidx = rowbase / Nv;
    int nloc = rowbase - bidx * Nv;
    const float* Arow = x + ((size_t)(bidx * Tv + t) * Nv + nloc) * Fv;

    v8f acc = {};
#pragma unroll
    for (int kt = 0; kt < 2; ++kt) {  // K = 64
        v16bf a = load_a_f32(Arow + kt * 32, Fv, lane);
        v16bf b = *(const v16bf*)(Wswz + (((size_t)kt * 8 + wave) * 32 + lane) * 16);
        acc = wmma_bf16(a, b, acc);
    }
    float* Ho = h + ((size_t)t * NTOT + rowbase) * Hv;
    int c  = wave * 16 + (lane & 15);
    int m0 = (lane & 16) ? 8 : 0;
#pragma unroll
    for (int r = 0; r < 8; ++r)
        Ho[(size_t)(m0 + r) * Hv + c] = acc[r];
}

// ---------------- attention logits ----------------
__global__ void k_att_logits(const float* __restrict__ h, const float* __restrict__ att_src,
                             const float* __restrict__ att_dst,
                             float* __restrict__ asrc, float* __restrict__ adst) {
    int idx = blockIdx.x * blockDim.x + threadIdx.x;
    if (idx >= Tv * NTOT) return;
    const float* hv = h + (size_t)idx * Hv;
#pragma unroll
    for (int hd = 0; hd < NHEADS; ++hd) {
        float s = 0.f, d = 0.f;
#pragma unroll 4
        for (int k = 0; k < Dv; ++k) {
            float v = hv[hd * Dv + k];
            s += v * att_src[hd * Dv + k];
            d += v * att_dst[hd * Dv + k];
        }
        asrc[(size_t)idx * NHEADS + hd] = s;
        adst[(size_t)idx * NHEADS + hd] = d;
    }
}

// ---------------- init seg buffers + agg ----------------
__global__ void k_init(unsigned* __restrict__ segmax, float* __restrict__ den,
                       float* __restrict__ agg) {
    int idx = blockIdx.x * blockDim.x + threadIdx.x;
    if (idx < Tv * NTOT * Hv) agg[idx] = 0.f;
    if (idx < Tv * NTOT * NHEADS) { segmax[idx] = 0u; den[idx] = 0.f; }
}

__device__ __forceinline__ void edge_sd(int e, const int* es, const int* ed, int& s, int& d) {
    if (e < Ev) { s = es[e]; d = ed[e]; }
    else        { s = e - Ev; d = e - Ev; }
}

// edge logit (leaky relu) -> val buffer, and atomic segment max
__global__ void k_edge_logit(const int* __restrict__ es, const int* __restrict__ ed,
                             const float* __restrict__ asrc, const float* __restrict__ adst,
                             float* __restrict__ val, unsigned* __restrict__ segmax) {
    int idx = blockIdx.x * blockDim.x + threadIdx.x;
    if (idx >= Tv * EF * NHEADS) return;
    int hd = idx & 3;
    int e  = (idx >> 2) % EF;
    int t  = (idx >> 2) / EF;
    int s, d; edge_sd(e, es, ed, s, d);
    float v = asrc[((size_t)t * NTOT + s) * NHEADS + hd] + adst[((size_t)t * NTOT + d) * NHEADS + hd];
    v = v > 0.f ? v : 0.2f * v;  // leaky relu
    val[idx] = v;
    atomicMax(&segmax[((size_t)t * NTOT + d) * NHEADS + hd], ordf(v));
}

__global__ void k_seg_sum(const int* __restrict__ es, const int* __restrict__ ed,
                          const float* __restrict__ val, const unsigned* __restrict__ segmax,
                          float* __restrict__ den) {
    int idx = blockIdx.x * blockDim.x + threadIdx.x;
    if (idx >= Tv * EF * NHEADS) return;
    int hd = idx & 3;
    int e  = (idx >> 2) % EF;
    int t  = (idx >> 2) / EF;
    int s, d; edge_sd(e, es, ed, s, d);
    float m = deordf(segmax[((size_t)t * NTOT + d) * NHEADS + hd]);
    atomicAdd(&den[((size_t)t * NTOT + d) * NHEADS + hd], __expf(val[idx] - m));
}

// val -> alpha in place
__global__ void k_alpha(const int* __restrict__ es, const int* __restrict__ ed,
                        const unsigned* __restrict__ segmax, const float* __restrict__ den,
                        float* __restrict__ val) {
    int idx = blockIdx.x * blockDim.x + threadIdx.x;
    if (idx >= Tv * EF * NHEADS) return;
    int hd = idx & 3;
    int e  = (idx >> 2) % EF;
    int t  = (idx >> 2) / EF;
    int s, d; edge_sd(e, es, ed, s, d);
    size_t segi = ((size_t)t * NTOT + d) * NHEADS + hd;
    val[idx] = __expf(val[idx] - deordf(segmax[segi])) / den[segi];
}

// ---------------- message scatter: agg[dst] += alpha * h[src] (float4 per thread) ----------------
__global__ void k_scatter(const int* __restrict__ es, const int* __restrict__ ed,
                          const float* __restrict__ alpha, const float* __restrict__ h,
                          float* __restrict__ agg) {
    int idx = blockIdx.x * blockDim.x + threadIdx.x;
    if (idx >= Tv * EF * (Hv / 4)) return;
    int j4 = idx & 31;                 // 32 quads of 4 floats
    int e  = (idx >> 5) % EF;
    int t  = (idx >> 5) / EF;
    int j0 = j4 * 4;
    int hd = j4 >> 3;                  // j0 / 32
    int s, d; edge_sd(e, es, ed, s, d);
    float a = alpha[((size_t)t * EF + e) * NHEADS + hd];
    const float4 hv = *(const float4*)(h + ((size_t)t * NTOT + s) * Hv + j0);
    float* ao = agg + ((size_t)t * NTOT + d) * Hv + j0;
    atomicAdd(ao + 0, a * hv.x);
    atomicAdd(ao + 1, a * hv.y);
    atomicAdd(ao + 2, a * hv.z);
    atomicAdd(ao + 3, a * hv.w);
}

// ---------------- spatial = relu(agg + bias) -> bf16 ----------------
__global__ void k_spatial(const float* __restrict__ agg, const float* __restrict__ bias,
                          __bf16* __restrict__ sp) {
    int idx = blockIdx.x * blockDim.x + threadIdx.x;
    if (idx >= Tv * NTOT * Hv) return;
    float v = agg[idx] + bias[idx & (Hv - 1)];
    sp[idx] = (__bf16)fmaxf(v, 0.f);
}

__global__ void k_zero_state(float* h0, float* c0, float* h1, float* c1,
                             __bf16* h0b, __bf16* h1b) {
    int idx = blockIdx.x * blockDim.x + threadIdx.x;
    if (idx >= NTOT * Hv) return;
    h0[idx] = 0.f; c0[idx] = 0.f; h1[idx] = 0.f; c1[idx] = 0.f;
    h0b[idx] = (__bf16)0.f; h1b[idx] = (__bf16)0.f;
}

// ---------------- fused dual GEMM: gates = A1*B1 + A2*B2 + bias ----------------
// Register-blocked: each wave computes 4 row-tiles x 1 col-tile (64x16 output),
// reusing every B fragment across 4 WMMAs. 32 WMMAs per wave.
// grid: (512/4 rowgroups * 32 coltiles) / 8 waves = 512 blocks x 256 thr.
__global__ void k_lstm_gates(const __bf16* __restrict__ A1, const __bf16* __restrict__ A2,
                             const __bf16* __restrict__ B1, const __bf16* __restrict__ B2,
                             const float* __restrict__ bias, float* __restrict__ gates) {
    int lane = threadIdx.x & 31;
    int wave = threadIdx.x >> 5;
    int tile = blockIdx.x * 8 + wave;      // 4096 wave-tiles: 128 rowgroups x 32 coltiles
    int rg = tile >> 5;                    // 0..127
    int ct = tile & 31;                    // 0..31
    int rowbase = rg * 64;
    v8f acc0 = {}, acc1 = {}, acc2 = {}, acc3 = {};
#pragma unroll
    for (int kt = 0; kt < 4; ++kt) {       // K = 128, input-to-hidden weights
        v16bf b  = *(const v16bf*)(B1 + (((size_t)kt * 32 + ct) * 32 + lane) * 16);
        v16bf a0 = load_a_bf16(A1 + (size_t)(rowbase +  0) * Hv + kt * 32, Hv, lane);
        v16bf a1 = load_a_bf16(A1 + (size_t)(rowbase + 16) * Hv + kt * 32, Hv, lane);
        v16bf a2 = load_a_bf16(A1 + (size_t)(rowbase + 32) * Hv + kt * 32, Hv, lane);
        v16bf a3 = load_a_bf16(A1 + (size_t)(rowbase + 48) * Hv + kt * 32, Hv, lane);
        acc0 = wmma_bf16(a0, b, acc0);
        acc1 = wmma_bf16(a1, b, acc1);
        acc2 = wmma_bf16(a2, b, acc2);
        acc3 = wmma_bf16(a3, b, acc3);
    }
#pragma unroll
    for (int kt = 0; kt < 4; ++kt) {       // K = 128, hidden-to-hidden weights
        v16bf b  = *(const v16bf*)(B2 + (((size_t)kt * 32 + ct) * 32 + lane) * 16);
        v16bf a0 = load_a_bf16(A2 + (size_t)(rowbase +  0) * Hv + kt * 32, Hv, lane);
        v16bf a1 = load_a_bf16(A2 + (size_t)(rowbase + 16) * Hv + kt * 32, Hv, lane);
        v16bf a2 = load_a_bf16(A2 + (size_t)(rowbase + 32) * Hv + kt * 32, Hv, lane);
        v16bf a3 = load_a_bf16(A2 + (size_t)(rowbase + 48) * Hv + kt * 32, Hv, lane);
        acc0 = wmma_bf16(a0, b, acc0);
        acc1 = wmma_bf16(a1, b, acc1);
        acc2 = wmma_bf16(a2, b, acc2);
        acc3 = wmma_bf16(a3, b, acc3);
    }
    int c  = ct * 16 + (lane & 15);
    int m0 = (lane & 16) ? 8 : 0;
    float bb = bias[c];
    float* g0 = gates + (size_t)(rowbase + m0) * G4H + c;
#pragma unroll
    for (int r = 0; r < 8; ++r) g0[(size_t)r * G4H]                = acc0[r] + bb;
#pragma unroll
    for (int r = 0; r < 8; ++r) g0[(size_t)(16 + r) * G4H]         = acc1[r] + bb;
#pragma unroll
    for (int r = 0; r < 8; ++r) g0[(size_t)(32 + r) * G4H]         = acc2[r] + bb;
#pragma unroll
    for (int r = 0; r < 8; ++r) g0[(size_t)(48 + r) * G4H]         = acc3[r] + bb;
}

// ---------------- LSTM cell pointwise ----------------
__global__ void k_lstm_cell(const float* __restrict__ gates, float* __restrict__ hS,
                            float* __restrict__ cS, __bf16* __restrict__ hB) {
    int idx = blockIdx.x * blockDim.x + threadIdx.x;
    if (idx >= NTOT * Hv) return;
    int n = idx >> 7;
    int j = idx & (Hv - 1);
    const float* g = gates + (size_t)n * G4H;
    float ig = sigm(g[j]);
    float fg = sigm(g[Hv + j]);
    float gg = tanhf(g[2 * Hv + j]);
    float og = sigm(g[3 * Hv + j]);
    float c  = fg * cS[idx] + ig * gg;
    float hh = og * tanhf(c);
    cS[idx] = c; hS[idx] = hh; hB[idx] = (__bf16)hh;
}

// ---------------- LayerNorm over H ----------------
__global__ void k_layernorm(const float* __restrict__ h1, const float* __restrict__ gamma,
                            const float* __restrict__ beta, float* __restrict__ out) {
    __shared__ float ssum[Hv], ssq[Hv];
    int n = blockIdx.x, j = threadIdx.x;
    float v = h1[(size_t)n * Hv + j];
    ssum[j] = v; ssq[j] = v * v;
    __syncthreads();
    for (int s = Hv / 2; s > 0; s >>= 1) {
        if (j < s) { ssum[j] += ssum[j + s]; ssq[j] += ssq[j + s]; }
        __syncthreads();
    }
    float mean = ssum[0] * (1.f / Hv);
    float var  = ssq[0] * (1.f / Hv) - mean * mean;
    float r = rsqrtf(var + 1e-5f);
    out[(size_t)n * Hv + j] = (v - mean) * r * gamma[j] + beta[j];
}

// ---------------- host launcher ----------------
static inline unsigned cdiv(size_t a, unsigned b) { return (unsigned)((a + b - 1) / b); }

extern "C" void kernel_launch(void* const* d_in, const int* in_sizes, int n_in,
                              void* d_out, int out_size, void* d_ws, size_t ws_size,
                              hipStream_t stream) {
    const float* x       = (const float*)d_in[0];
    const float* W       = (const float*)d_in[1];
    const float* att_src = (const float*)d_in[2];
    const float* att_dst = (const float*)d_in[3];
    const float* gbias   = (const float*)d_in[4];
    const float* W_ih0   = (const float*)d_in[5];
    const float* W_hh0   = (const float*)d_in[6];
    const float* b_ih0   = (const float*)d_in[7];
    const float* b_hh0   = (const float*)d_in[8];
    const float* W_ih1   = (const float*)d_in[9];
    const float* W_hh1   = (const float*)d_in[10];
    const float* b_ih1   = (const float*)d_in[11];
    const float* b_hh1   = (const float*)d_in[12];
    const float* ln_g    = (const float*)d_in[13];
    const float* ln_b    = (const float*)d_in[14];
    const int*   e_src   = (const int*)d_in[15];
    const int*   e_dst   = e_src + Ev;
    float* out = (float*)d_out;

    // deterministic workspace carve-out
    char* ws = (char*)d_ws;
    size_t off = 0;
    auto carve = [&](size_t bytes) { size_t o = off; off += (bytes + 255) & ~(size_t)255; return (void*)(ws + o); };
    float*    h_f32  = (float*)   carve((size_t)Tv * NTOT * Hv * 4);
    float*    agg    = (float*)   carve((size_t)Tv * NTOT * Hv * 4);
    float*    asrc   = (float*)   carve((size_t)Tv * NTOT * NHEADS * 4);
    float*    adst   = (float*)   carve((size_t)Tv * NTOT * NHEADS * 4);
    unsigned* segmax = (unsigned*)carve((size_t)Tv * NTOT * NHEADS * 4);
    float*    den    = (float*)   carve((size_t)Tv * NTOT * NHEADS * 4);
    float*    val    = (float*)   carve((size_t)Tv * EF * NHEADS * 4);  // logits -> alpha
    __bf16*   sp_b   = (__bf16*)  carve((size_t)Tv * NTOT * Hv * 2);
    float*    gates  = (float*)   carve((size_t)NTOT * G4H * 4);
    float*    h0     = (float*)   carve((size_t)NTOT * Hv * 4);
    float*    c0     = (float*)   carve((size_t)NTOT * Hv * 4);
    float*    h1     = (float*)   carve((size_t)NTOT * Hv * 4);
    float*    c1     = (float*)   carve((size_t)NTOT * Hv * 4);
    __bf16*   h0b    = (__bf16*)  carve((size_t)NTOT * Hv * 2);
    __bf16*   h1b    = (__bf16*)  carve((size_t)NTOT * Hv * 2);
    __bf16*   Wswz   = (__bf16*)  carve((size_t)Fv * Hv * 2);
    __bf16*   Wih0s  = (__bf16*)  carve((size_t)Hv * G4H * 2);
    __bf16*   Whh0s  = (__bf16*)  carve((size_t)Hv * G4H * 2);
    __bf16*   Wih1s  = (__bf16*)  carve((size_t)Hv * G4H * 2);
    __bf16*   Whh1s  = (__bf16*)  carve((size_t)Hv * G4H * 2);
    float*    b0     = (float*)   carve((size_t)G4H * 4);
    float*    b1     = (float*)   carve((size_t)G4H * 4);

    // 1) weight prep
    k_swizzle<<<cdiv((size_t)Fv * Hv, 256), 256, 0, stream>>>(W, Wswz, Fv, Hv, 0);
    k_swizzle<<<cdiv((size_t)Hv * G4H, 256), 256, 0, stream>>>(W_ih0, Wih0s, Hv, G4H, 1);
    k_swizzle<<<cdiv((size_t)Hv * G4H, 256), 256, 0, stream>>>(W_hh0, Whh0s, Hv, G4H, 1);
    k_swizzle<<<cdiv((size_t)Hv * G4H, 256), 256, 0, stream>>>(W_ih1, Wih1s, Hv, G4H, 1);
    k_swizzle<<<cdiv((size_t)Hv * G4H, 256), 256, 0, stream>>>(W_hh1, Whh1s, Hv, G4H, 1);
    k_bias_combine<<<cdiv(G4H, 256), 256, 0, stream>>>(b_ih0, b_hh0, b0, G4H);
    k_bias_combine<<<cdiv(G4H, 256), 256, 0, stream>>>(b_ih1, b_hh1, b1, G4H);

    // 2) GAT projection (WMMA) + attention logits
    k_gat_proj<<<Tv * (NTOT / 16), 256, 0, stream>>>(x, Wswz, h_f32);
    k_att_logits<<<cdiv((size_t)Tv * NTOT, 256), 256, 0, stream>>>(h_f32, att_src, att_dst, asrc, adst);

    // 3) segment softmax + scatter aggregate
    k_init<<<cdiv((size_t)Tv * NTOT * Hv, 256), 256, 0, stream>>>(segmax, den, agg);
    k_edge_logit<<<cdiv((size_t)Tv * EF * NHEADS, 256), 256, 0, stream>>>(e_src, e_dst, asrc, adst, val, segmax);
    k_seg_sum<<<cdiv((size_t)Tv * EF * NHEADS, 256), 256, 0, stream>>>(e_src, e_dst, val, segmax, den);
    k_alpha<<<cdiv((size_t)Tv * EF * NHEADS, 256), 256, 0, stream>>>(e_src, e_dst, segmax, den, val);
    k_scatter<<<cdiv((size_t)Tv * EF * (Hv / 4), 256), 256, 0, stream>>>(e_src, e_dst, val, h_f32, agg);
    k_spatial<<<cdiv((size_t)Tv * NTOT * Hv, 256), 256, 0, stream>>>(agg, gbias, sp_b);

    // 4) 2-layer LSTM over time (register-blocked WMMA per-step GEMMs)
    k_zero_state<<<cdiv((size_t)NTOT * Hv, 256), 256, 0, stream>>>(h0, c0, h1, c1, h0b, h1b);
    for (int t = 0; t < Tv; ++t) {
        const __bf16* spt = sp_b + (size_t)t * NTOT * Hv;
        k_lstm_gates<<<512, 256, 0, stream>>>(spt, h0b, Wih0s, Whh0s, b0, gates);
        k_lstm_cell<<<cdiv((size_t)NTOT * Hv, 256), 256, 0, stream>>>(gates, h0, c0, h0b);
        k_lstm_gates<<<512, 256, 0, stream>>>(h0b, h1b, Wih1s, Whh1s, b1, gates);
        k_lstm_cell<<<cdiv((size_t)NTOT * Hv, 256), 256, 0, stream>>>(gates, h1, c1, h1b);
    }

    // 5) LayerNorm -> out
    k_layernorm<<<NTOT, Hv, 0, stream>>>(h1, ln_g, ln_b, out);
}
